// ORNet_43370579755058
// MI455X (gfx1250) — matrix operand order repaired and checked by reading, not compiled, not run on hardware
//
#include <hip/hip_runtime.h>

// ---------------------------------------------------------------------------
// ORNet fused kernel for MI455X (gfx1250, wave32, WMMA).
//   z = x @ W^T            (M=4096, N=1024, K=2048, fp32 in)
//   t_j = sigmoid(z + bias_j), j=0..6
//   out[...,0]=t0, out[...,j]=t_j - t_{j-1}, out[...,7]=t6  -> [4096, 8192] f32
//
// Precision: split-bf16 (bf16x3) GEMM: x=xh+xl, W=wh+wl in bf16;
//   acc += ah*bh + ah*bl + al*bh  accumulated in f32 via v_wmma_f32_16x16x32_bf16.
// Roofline: out write (134MB) dominates -> ~7.5us memory floor @23.3TB/s;
//   3x17.2GFLOP of bf16 WMMA keeps the XDL pipe near that floor, conversion
//   VALU co-executes under the WMMA shadow.
// ---------------------------------------------------------------------------

typedef __bf16 bf16_t;
typedef bf16_t v16bf __attribute__((ext_vector_type(16)));
typedef bf16_t v8bf  __attribute__((ext_vector_type(8)));
typedef bf16_t v4bf  __attribute__((ext_vector_type(4)));
typedef float  v8f   __attribute__((ext_vector_type(8)));
typedef float  v4f   __attribute__((ext_vector_type(4)));

#define KDIM 2048
#define NCLS 1024
#define MTOT 4096
#define SM1  7

#define MBLK 128
#define NBLK 128
#define KT   32
#define NKT  (KDIM / KT)
#define LDK  40   // padded LDS row stride in bf16 elements (80 bytes)

__device__ __forceinline__ float fast_sigmoid(float x) {
  // sigmoid(x) = 1 / (1 + exp2(-x * log2(e)))
  float e = __builtin_amdgcn_exp2f(x * -1.44269504088896341f);
  return __builtin_amdgcn_rcpf(1.0f + e);
}

__global__ __launch_bounds__(256) void ornet_wmma_kernel(
    const float* __restrict__ X,     // [4096, 2048]
    const float* __restrict__ W,     // [1024, 2048]
    const float* __restrict__ BIAS,  // [1024, 7]
    float* __restrict__ OUT)         // [4096, 8192]
{
  __shared__ bf16_t Ah[2][MBLK * LDK];
  __shared__ bf16_t Al[2][MBLK * LDK];
  __shared__ bf16_t Bh[2][NBLK * LDK];
  __shared__ bf16_t Bl[2][NBLK * LDK];

  const int tid  = threadIdx.x;
  const int lane = tid & 31;
  const int wave = tid >> 5;       // 0..7
  const int wm   = wave >> 1;      // 0..3 : wave row  (32 M-rows each)
  const int wn   = wave & 1;       // 0..1 : wave col  (64 N-cols each)
  const int lr   = lane & 15;      // row/col within a 16-wide fragment
  const int lh   = lane >> 4;      // fragment half select

  const int m0 = blockIdx.x * MBLK;
  const int n0 = blockIdx.y * NBLK;

  // ---- global->reg copy mapping: 256 threads, float4 per load ----
  const int crow = tid >> 3;          // 0..31 (row within a 32-row pass)
  const int ccol = (tid & 7) << 2;    // 0,4,...,28 (K element offset)

  const float* ax = X + (size_t)(m0 + crow) * KDIM + ccol;
  const float* bw = W + (size_t)(n0 + crow) * KDIM + ccol;

  v4f ra[4], rb[4];

  auto gload = [&](int kt) {
    const float* pa = ax + kt * KT;
    const float* pb = bw + kt * KT;
#pragma unroll
    for (int p = 0; p < 4; ++p) {
      ra[p] = *(const v4f*)(pa + (size_t)p * 32 * KDIM);
      rb[p] = *(const v4f*)(pb + (size_t)p * 32 * KDIM);
    }
  };

  // whole-vector split: 2x v_cvt_pk_bf16_f32 (hi) + 4x fma_mix (residual)
  // + 2x v_cvt_pk_bf16_f32 (lo); no per-element half-register moves.
  auto split_store = [&](v4f f, bf16_t* ph, bf16_t* pl) {
    v4bf h  = __builtin_convertvector(f, v4bf);
    v4f  hf = __builtin_convertvector(h, v4f);
    v4f  d  = f - hf;
    v4bf l  = __builtin_convertvector(d, v4bf);
    *(v4bf*)ph = h;
    *(v4bf*)pl = l;
  };

  auto lstore = [&](int buf) {
#pragma unroll
    for (int p = 0; p < 4; ++p) {
      const int row = p * 32 + crow;
      split_store(ra[p], &Ah[buf][row * LDK + ccol], &Al[buf][row * LDK + ccol]);
      split_store(rb[p], &Bh[buf][row * LDK + ccol], &Bl[buf][row * LDK + ccol]);
    }
  };

  // A fragment (16x32 bf16, M x K): lane holds row M=lr,
  //   K chunks {8*lh .. 8*lh+7} and {16+8*lh .. 16+8*lh+7}
  auto loadA = [&](int buf, int mf, v16bf& hi, v16bf& lo) {
    const int row = wm * 32 + mf * 16 + lr;
    const bf16_t* ph = &Ah[buf][row * LDK];
    const bf16_t* pl = &Al[buf][row * LDK];
    v8bf h0 = *(const v8bf*)(ph + 8 * lh);
    v8bf h1 = *(const v8bf*)(ph + 16 + 8 * lh);
    v8bf l0 = *(const v8bf*)(pl + 8 * lh);
    v8bf l1 = *(const v8bf*)(pl + 16 + 8 * lh);
    hi = __builtin_shufflevector(h0, h1, 0,1,2,3,4,5,6,7,8,9,10,11,12,13,14,15);
    lo = __builtin_shufflevector(l0, l1, 0,1,2,3,4,5,6,7,8,9,10,11,12,13,14,15);
  };

  // B fragment (32x16 bf16, K x N): lane holds col N=lr,
  //   K chunks {8*lh .. 8*lh+7} and {16+8*lh .. 16+8*lh+7}
  auto loadB = [&](int buf, int nf, v16bf& hi, v16bf& lo) {
    const int row = wn * 64 + nf * 16 + lr;
    const bf16_t* ph = &Bh[buf][row * LDK + 8 * lh];
    const bf16_t* pl = &Bl[buf][row * LDK + 8 * lh];
    v8bf h0 = *(const v8bf*)(ph);
    v8bf h1 = *(const v8bf*)(ph + 16);
    v8bf l0 = *(const v8bf*)(pl);
    v8bf l1 = *(const v8bf*)(pl + 16);
    hi = __builtin_shufflevector(h0, h1, 0,1,2,3,4,5,6,7,8,9,10,11,12,13,14,15);
    lo = __builtin_shufflevector(l0, l1, 0,1,2,3,4,5,6,7,8,9,10,11,12,13,14,15);
  };

  v8f acc[2][4];
#pragma unroll
  for (int i = 0; i < 2; ++i)
#pragma unroll
    for (int j = 0; j < 4; ++j)
      acc[i][j] = (v8f)(0.0f);

  auto compute = [&](int buf) {
    v16bf ah[2], al[2];
    loadA(buf, 0, ah[0], al[0]);
    loadA(buf, 1, ah[1], al[1]);
#pragma unroll
    for (int nf = 0; nf < 4; ++nf) {
      v16bf bh, bl;
      loadB(buf, nf, bh, bl);
#pragma unroll
      for (int mf = 0; mf < 2; ++mf) {
        acc[mf][nf] = __builtin_amdgcn_wmma_f32_16x16x32_bf16(
            false, ah[mf], false, bh, (short)0, acc[mf][nf], false, false);
        acc[mf][nf] = __builtin_amdgcn_wmma_f32_16x16x32_bf16(
            false, ah[mf], false, bl, (short)0, acc[mf][nf], false, false);
        acc[mf][nf] = __builtin_amdgcn_wmma_f32_16x16x32_bf16(
            false, al[mf], false, bh, (short)0, acc[mf][nf], false, false);
      }
    }
  };

  // ---- K loop: double-buffered LDS, one barrier per K-step ----
  gload(0);
  lstore(0);
  int buf = 0;
  for (int kt = 0; kt < NKT; ++kt) {
    if (kt + 1 < NKT) gload(kt + 1);
    __syncthreads();
    compute(buf);
    if (kt + 1 < NKT) lstore(buf ^ 1);
    buf ^= 1;
  }

  // ---- epilogue: ordinal-regression head, fully coalesced f32 stores ----
  const int nbase = n0 + wn * 64;
#pragma unroll
  for (int nf = 0; nf < 4; ++nf) {
    const int n = nbase + nf * 16 + lr;
    const float* bp = BIAS + (size_t)n * SM1;
    float b[SM1];
#pragma unroll
    for (int j = 0; j < SM1; ++j) b[j] = bp[j];

#pragma unroll
    for (int mf = 0; mf < 2; ++mf) {
      const int mrow = m0 + wm * 32 + mf * 16 + 8 * lh;
#pragma unroll
      for (int r = 0; r < 8; ++r) {
        const float z = acc[mf][nf][r];
        float t[SM1];
#pragma unroll
        for (int j = 0; j < SM1; ++j) t[j] = fast_sigmoid(z + b[j]);
        v4f o0 = {t[0], t[1] - t[0], t[2] - t[1], t[3] - t[2]};
        v4f o1 = {t[4] - t[3], t[5] - t[4], t[6] - t[5], t[6]};
        float* op = OUT + (size_t)(mrow + r) * (NCLS * 8) + (size_t)n * 8;
        *(v4f*)op       = o0;
        *(v4f*)(op + 4) = o1;
      }
    }
  }
}

extern "C" void kernel_launch(void* const* d_in, const int* in_sizes, int n_in,
                              void* d_out, int out_size, void* d_ws, size_t ws_size,
                              hipStream_t stream) {
  (void)in_sizes; (void)n_in; (void)d_ws; (void)ws_size; (void)out_size;
  const float* x  = (const float*)d_in[0];  // [4096, 2048]
  const float* w  = (const float*)d_in[1];  // [1024, 2048]
  const float* ob = (const float*)d_in[2];  // [1024, 7]
  float* out = (float*)d_out;               // [4096, 8192]

  dim3 grid(MTOT / MBLK, NCLS / NBLK);      // 32 x 8 = 256 workgroups
  ornet_wmma_kernel<<<grid, 256, 0, stream>>>(x, w, ob, out);
}
